// Encoder_12876311953615
// MI455X (gfx1250) — compile-verified, matrix operand order checked by reading, not run on hardware
//
#include <hip/hip_runtime.h>
#include <hip/hip_bf16.h>

typedef __attribute__((ext_vector_type(16))) __bf16 v16bf;
typedef __attribute__((ext_vector_type(8)))  __bf16 v8bf;
typedef __attribute__((ext_vector_type(8)))  float  v8f;

#define B_ 128
#define T_ 1024
#define E_ 256
#define H_ 256

__device__ __forceinline__ float sigmoidf_(float x) {
    return 1.0f / (1.0f + __expf(-x));
}

#define SHUF16(lo, hi) __builtin_shufflevector(lo, hi, 0,1,2,3,4,5,6,7,8,9,10,11,12,13,14,15)
#define LOADF16(ptr, ko) SHUF16(*(const v8bf*)((ptr) + (ko)), *(const v8bf*)((ptr) + (ko) + 16))

// ---------------- prep kernels (one-shot, parallel) ----------------

// WAt[dir][n][k], n = gate*256 + hcol in [0,768), k in [0,512):
//   k <  256 -> Wx[g][k][hcol]
//   k >= 256 -> (g<2 ? Wh[g][k-256][hcol] : 0)   (candidate Wh block never read)
__global__ void prep_wa(const float* __restrict__ Wx_r, const float* __restrict__ Wh_r,
                        const float* __restrict__ Wx_l, const float* __restrict__ Wh_l,
                        __bf16* __restrict__ WAt) {
    int id = blockIdx.x * blockDim.x + threadIdx.x;
    if (id >= 2 * 768 * 512) return;
    int dir = id / (768 * 512);
    int r   = id % (768 * 512);
    int n   = r / 512;
    int k   = r % 512;
    int g   = n >> 8;
    int hc  = n & 255;
    const float* Wx = dir ? Wx_l : Wx_r;
    const float* Wh = dir ? Wh_l : Wh_r;
    float v;
    if (k < 256)      v = Wx[((size_t)g * 256 + k) * 256 + hc];
    else if (g < 2)   v = Wh[((size_t)g * 256 + (k - 256)) * 256 + hc];
    else              v = 0.0f;
    WAt[id] = (__bf16)v;
}

// Wh2t[dir][n][k] = Wh[dir][2][k][n]
__global__ void prep_wh2(const float* __restrict__ Wh_r, const float* __restrict__ Wh_l,
                         __bf16* __restrict__ Wh2t) {
    int id = blockIdx.x * blockDim.x + threadIdx.x;
    if (id >= 2 * 256 * 256) return;
    int dir = id >> 16;
    int r   = id & 65535;
    int n   = r >> 8;
    int k   = r & 255;
    const float* Wh = dir ? Wh_l : Wh_r;
    Wh2t[id] = (__bf16)Wh[((size_t)(2 * 256) + k) * 256 + n];
}

// bias[dir][g*256+h] = bx[g][h] + bh[g][h]
__global__ void prep_bias(const float* __restrict__ bx_r, const float* __restrict__ bh_r,
                          const float* __restrict__ bx_l, const float* __restrict__ bh_l,
                          float* __restrict__ bia) {
    int id = blockIdx.x * blockDim.x + threadIdx.x;
    if (id >= 2 * 768) return;
    int dir = id / 768;
    int i   = id % 768;
    bia[id] = dir ? (bx_l[i] + bh_l[i]) : (bx_r[i] + bh_r[i]);
}

// Embedding gather + fp32->bf16 (write-once, non-temporal)
__global__ void gather_emb(const int* __restrict__ x, const float* __restrict__ emb,
                           __bf16* __restrict__ Xbf) {
    size_t id = (size_t)blockIdx.x * blockDim.x + threadIdx.x;
    int    e  = (int)(id & 255);
    size_t bt = id >> 8;
    int tok   = x[bt];
    float v   = (tok == 0) ? 0.0f : emb[(size_t)tok * 256 + e];
    __builtin_nontemporal_store((__bf16)v, Xbf + id);
}

// ---------------- persistent recurrence kernel ----------------
// grid = 16 blocks: blockIdx = dir*8 + btile. block = 256 threads = 8 waves.
// Wave w owns N-tiles {w, w+8, ..., w+40}: hidden columns {16w+i, 16(w+8)+i}
// for z, r, candidate all live in the same lanes -> no cross-lane traffic.
__global__ __launch_bounds__(256, 1) void gru_recurrence(
    const __bf16* __restrict__ WAt, const __bf16* __restrict__ Wh2t,
    const float* __restrict__ biasc, const __bf16* __restrict__ Xbf,
    float* __restrict__ out) {
    __shared__ __bf16 Xld[2][16][264];  // double-buffered x_t tile (async-filled)
    __shared__ __bf16 Hld[16][264];     // hidden state (bf16)
    __shared__ __bf16 RHld[16][264];    // r .* h

    const int dir   = blockIdx.x >> 3;
    const int btile = blockIdx.x & 7;
    const int tid   = threadIdx.x;
    const int w     = tid >> 5;
    const int lane  = tid & 31;
    const int l15   = lane & 15;
    const int hi8   = (lane >> 4) << 3;  // K half-select; also C-frag row base

    // x-tile loader mapping: 16 rows x 16 segments of 16 bf16 (32 B)
    const int xrow   = tid >> 4;
    const int xseg   = tid & 15;
    const int outb_x = btile * 16 + xrow;
    const int srcb_x = dir ? (B_ - 1 - outb_x) : outb_x;  // left RNN: batch-reversed stream
    const __bf16* xsrc_base = Xbf + ((size_t)srcb_x * T_) * E_ + xseg * 16;

    // LDS destination byte addresses (flat addr low 32 bits == LDS offset)
    const unsigned xlds0 = (unsigned)(size_t)&Xld[0][xrow][xseg * 16];
    const unsigned xlds1 = (unsigned)(size_t)&Xld[1][xrow][xseg * 16];

    // prologue: async-load x_0 into buffer 0; zero hidden state
    asm volatile("global_load_async_to_lds_b128 %0, %1, off\n\t"
                 "global_load_async_to_lds_b128 %0, %1, off offset:16"
                 :: "v"(xlds0), "v"(xsrc_base) : "memory");
    {
        uint4  z4 = make_uint4(0u, 0u, 0u, 0u);
        uint4* d  = (uint4*)&Hld[xrow][xseg * 16];
        d[0] = z4; d[1] = z4;
    }

    // per-lane column metadata
    const float* bvec = biasc + dir * 768;
    const int hc0 = 16 * w + l15;
    const int hc1 = 16 * (w + 8) + l15;
    const float bz0 = bvec[hc0],       bz1 = bvec[hc1];
    const float br0 = bvec[256 + hc0], br1 = bvec[256 + hc1];
    const float bg0 = bvec[512 + hc0], bg1 = bvec[512 + hc1];

    // phase-A B-fragment base pointers (streamed from L2 each step)
    const __bf16* wa_base = WAt + (size_t)dir * 768 * 512;
    const __bf16* wb[6];
#pragma unroll
    for (int j = 0; j < 6; ++j)
        wb[j] = wa_base + (size_t)((w + 8 * j) * 16 + l15) * 512;

    // phase-B weights: loop-invariant, resident in VGPRs (16 x v16bf = 128 regs)
    const __bf16* w2b0 = Wh2t + (size_t)dir * 256 * 256 + (size_t)hc0 * 256;
    const __bf16* w2b1 = Wh2t + (size_t)dir * 256 * 256 + (size_t)hc1 * 256;
    v16bf w2f0[8], w2f1[8];
#pragma unroll
    for (int kc = 0; kc < 8; ++kc) {
        const int ko = 32 * kc + hi8;
        w2f0[kc] = LOADF16(w2b0, ko);
        w2f1[kc] = LOADF16(w2b1, ko);
    }

    const __bf16* xrow_base0 = &Xld[0][l15][0];
    const __bf16* xrow_base1 = &Xld[1][l15][0];
    const __bf16* hrow_base  = &Hld[l15][0];
    const __bf16* rhrow_base = &RHld[l15][0];

    float* out_base = out + (size_t)(btile * 16) * T_ * 512 + (size_t)dir * 256;

    asm volatile("s_wait_asynccnt 0x0" ::: "memory");
    __syncthreads();

    for (int t = 0; t < T_; ++t) {
        // kick off async fill of x_{t+1}; it lands by the end of this step
        if (t + 1 < T_) {
            const unsigned ldsa = ((t + 1) & 1) ? xlds1 : xlds0;
            const __bf16*  g    = xsrc_base + (size_t)(t + 1) * E_;
            asm volatile("global_load_async_to_lds_b128 %0, %1, off\n\t"
                         "global_load_async_to_lds_b128 %0, %1, off offset:16"
                         :: "v"(ldsa), "v"(g) : "memory");
        }
        const __bf16* xb = (t & 1) ? xrow_base1 : xrow_base0;

        v8f acc[6];
#pragma unroll
        for (int j = 0; j < 6; ++j)
#pragma unroll
            for (int i = 0; i < 8; ++i) acc[j][i] = 0.0f;

        // ---- Phase A2 first (on the h critical path): h @ Wh, z/r tiles only ----
        // explicit double-buffered B fragments keep the weight stream pipelined
        {
            v16bf b2[2][4];
#pragma unroll
            for (int j = 0; j < 4; ++j) b2[0][j] = LOADF16(wb[j], 256 + hi8);
#pragma unroll
            for (int kc = 0; kc < 8; ++kc) {
                const int cur = kc & 1;
                if (kc < 7) {
                    const int kn = 256 + 32 * (kc + 1) + hi8;
#pragma unroll
                    for (int j = 0; j < 4; ++j) b2[cur ^ 1][j] = LOADF16(wb[j], kn);
                }
                v16bf af = LOADF16(hrow_base, 32 * kc + hi8);
#pragma unroll
                for (int j = 0; j < 4; ++j)
                    acc[j] = __builtin_amdgcn_wmma_f32_16x16x32_bf16(
                        false, af, false, b2[cur][j], (short)0, acc[j], false, false);
            }
        }

        // ---- Phase A1 (off critical path): x_t @ Wx, all 6 N-tiles ----
        {
            v16bf b1[2][6];
#pragma unroll
            for (int j = 0; j < 6; ++j) b1[0][j] = LOADF16(wb[j], hi8);
#pragma unroll
            for (int kc = 0; kc < 8; ++kc) {
                const int cur = kc & 1;
                if (kc < 7) {
                    const int kn = 32 * (kc + 1) + hi8;
#pragma unroll
                    for (int j = 0; j < 6; ++j) b1[cur ^ 1][j] = LOADF16(wb[j], kn);
                }
                v16bf af = LOADF16(xb, 32 * kc + hi8);
#pragma unroll
                for (int j = 0; j < 6; ++j)
                    acc[j] = __builtin_amdgcn_wmma_f32_16x16x32_bf16(
                        false, af, false, b1[cur][j], (short)0, acc[j], false, false);
            }
        }

        // ---- z, r activations; stash r .* h; bias the candidate ----
        float zv0[8], zv1[8];
#pragma unroll
        for (int v = 0; v < 8; ++v) {
            zv0[v]   = sigmoidf_(acc[0][v] + bz0);
            zv1[v]   = sigmoidf_(acc[1][v] + bz1);
            float r0 = sigmoidf_(acc[2][v] + br0);
            float r1 = sigmoidf_(acc[3][v] + br1);
            int   m  = hi8 + v;
            float h0 = (float)Hld[m][hc0];
            float h1 = (float)Hld[m][hc1];
            RHld[m][hc0] = (__bf16)(r0 * h0);
            RHld[m][hc1] = (__bf16)(r1 * h1);
            acc[4][v] += bg0;
            acc[5][v] += bg1;
        }
        __syncthreads();

        // ---- Phase B: (r.*h) @ Wh2, weights already in registers ----
#pragma unroll
        for (int kc = 0; kc < 8; ++kc) {
            v16bf af = LOADF16(rhrow_base, 32 * kc + hi8);
            acc[4] = __builtin_amdgcn_wmma_f32_16x16x32_bf16(
                false, af, false, w2f0[kc], (short)0, acc[4], false, false);
            acc[5] = __builtin_amdgcn_wmma_f32_16x16x32_bf16(
                false, af, false, w2f1[kc], (short)0, acc[5], false, false);
        }

        // ---- finalize: h' = (1-z)*h + z*tanh(a_cand); NT-store output ----
#pragma unroll
        for (int v = 0; v < 8; ++v) {
            int   m   = hi8 + v;
            float h0  = (float)Hld[m][hc0];
            float h1  = (float)Hld[m][hc1];
            float ht0 = tanhf(acc[4][v]);
            float ht1 = tanhf(acc[5][v]);
            float hn0 = (1.0f - zv0[v]) * h0 + zv0[v] * ht0;
            float hn1 = (1.0f - zv1[v]) * h1 + zv1[v] * ht1;
            float* o  = out_base + ((size_t)m * T_ + t) * 512;
            __builtin_nontemporal_store(hn0, o + hc0);
            __builtin_nontemporal_store(hn1, o + hc1);
            // (m,hc) is touched only by this lane this step; next-step readers
            // are guarded by the end-of-step barrier below
            Hld[m][hc0] = (__bf16)hn0;
            Hld[m][hc1] = (__bf16)hn1;
        }

        // x_{t+1} async fill must have landed; h writes must be visible
        if (t + 1 < T_) asm volatile("s_wait_asynccnt 0x0" ::: "memory");
        __syncthreads();
    }
}

// ---------------- launcher ----------------
extern "C" void kernel_launch(void* const* d_in, const int* in_sizes, int n_in,
                              void* d_out, int out_size, void* d_ws, size_t ws_size,
                              hipStream_t stream) {
    (void)in_sizes; (void)n_in; (void)out_size; (void)ws_size;
    const int*   x    = (const int*)d_in[0];
    const float* emb  = (const float*)d_in[1];
    const float* Wx_r = (const float*)d_in[2];
    const float* bx_r = (const float*)d_in[3];
    const float* Wh_r = (const float*)d_in[4];
    const float* bh_r = (const float*)d_in[5];
    const float* Wx_l = (const float*)d_in[6];
    const float* bx_l = (const float*)d_in[7];
    const float* Wh_l = (const float*)d_in[8];
    const float* bh_l = (const float*)d_in[9];
    float* out = (float*)d_out;

    char* ws = (char*)d_ws;
    __bf16* WAt  = (__bf16*)ws;                              // 1,572,864 B
    __bf16* Wh2t = (__bf16*)(ws + 1572864);                  //   262,144 B
    float*  bia  = (float*) (ws + 1572864 + 262144);         //     6,144 B (pad 8K)
    __bf16* Xbf  = (__bf16*)(ws + 1572864 + 262144 + 8192);  // 64 MiB

    prep_wa  <<<(2 * 768 * 512 + 255) / 256, 256, 0, stream>>>(Wx_r, Wh_r, Wx_l, Wh_l, WAt);
    prep_wh2 <<<(2 * 256 * 256 + 255) / 256, 256, 0, stream>>>(Wh_r, Wh_l, Wh2t);
    prep_bias<<<(2 * 768       + 255) / 256, 256, 0, stream>>>(bx_r, bh_r, bx_l, bh_l, bia);
    gather_emb<<<(B_ * T_ * E_) / 256, 256, 0, stream>>>(x, emb, Xbf);
    gru_recurrence<<<16, 256, 0, stream>>>(WAt, Wh2t, bia, Xbf, out);
}
